// EfficientAttention_72258529787930
// MI455X (gfx1250) — matrix operand order repaired and checked by reading, not compiled, not run on hardware
//
#include <hip/hip_runtime.h>
#include <hip/hip_bf16.h>

// ---------------------------------------------------------------------------
// MI455X (gfx1250) linear-attention pipeline, bf16 WMMA with f32 accumulate.
// ---------------------------------------------------------------------------

typedef __attribute__((ext_vector_type(16))) __bf16 v16bf;
typedef __attribute__((ext_vector_type(8)))  float  v8f;
typedef __attribute__((ext_vector_type(2)))  __bf16 bf16x2;

#define WMMA_BF16(A, B, C) \
    __builtin_amdgcn_wmma_f32_16x16x32_bf16(false, (A), false, (B), (short)0, (C), false, false)

union FragU { v16bf v; uint4 q[2]; };

// Pack two f32 -> one dword of bf16 pairs (v_cvt_pk_bf16_f32).
__device__ __forceinline__ unsigned int pkbf(float a, float b) {
    union { bf16x2 v; unsigned int u; } r;
    r.v.x = (__bf16)a; r.v.y = (__bf16)b;
    return r.u;
}

// A-matrix 16x32 bf16 fragment (ISA 7.12.2): lane&15 = M row, kb=(lane>>4)*8.
// VGPR0..3 hold k = kb..kb+7, VGPR4..7 hold k = 16+kb..16+kb+7 (b128 pairs).
template <typename PT>
__device__ __forceinline__ v16bf load_frag_A(PT base, int row, int rowStride, int lane) {
    FragU f;
    PT p = base + (size_t)(row + (lane & 15)) * rowStride;
    const int kb = (lane >> 4) * 8;
    f.q[0] = *(const uint4*)(p + kb);
    f.q[1] = *(const uint4*)(p + 16 + kb);
    return f.v;
}

// B-matrix 32x16 bf16 fragment: lane&15 = N col, k = kb2..kb2+7 and 16+kb2..16+kb2+7.
template <typename PT>
__device__ __forceinline__ v16bf load_frag_B(PT base, int col, int rowStride, int lane) {
    FragU f;
    PT p = base + (size_t)(col + (lane & 15)) * rowStride;
    const int kb2 = (lane >> 4) * 8;
    f.q[0] = *(const uint4*)(p + kb2);
    f.q[1] = *(const uint4*)(p + kb2 + 16);
    return f.v;
}

// ---------------------------------------------------------------------------
// Kernel 1: fused QKV GEMM.  out[256,8192] = W[256,256] @ X[256,8192] + b.
// grid = (tok/128, 6, batch); WG tile 128x128; 8 waves as 4(M)x2(N).
// ---------------------------------------------------------------------------
__global__ __launch_bounds__(256) void qkv_kernel(
    const float* __restrict__ x,
    const float* __restrict__ Wq, const float* __restrict__ bq,
    const float* __restrict__ Wk, const float* __restrict__ bk,
    const float* __restrict__ Wv, const float* __restrict__ bv,
    float* __restrict__ Qf, float* __restrict__ Kf, __bf16* __restrict__ Vb)
{
    __shared__ __bf16 As[128 * 32];   // weight tile, k-contiguous rows
    __shared__ __bf16 Bs[128 * 32];   // X tile, n-major (k-contiguous)

    const int t = threadIdx.x;
    const int lane = t & 31, wave = t >> 5;
    const int which = blockIdx.y >> 1;              // 0=Q 1=K 2=V
    const int mBase = (blockIdx.y & 1) * 128;
    const int nBase = blockIdx.x * 128;
    const int batch = blockIdx.z;

    const float* W  = (which == 0) ? Wq : (which == 1) ? Wk : Wv;
    const float* bs = (which == 0) ? bq : (which == 1) ? bk : bv;
    const float* X  = x + (size_t)batch * 256 * 8192;  // reshape view [256][8192]

    const int mOff = (wave & 3) * 32;
    const int nOff = (wave >> 2) * 64;

    v8f acc[2][4] = {};

    for (int kk = 0; kk < 256; kk += 32) {
        __syncthreads();
        {   // stage A: W[mBase..+127][kk..+31] -> As[r][k] as packed bf16
            const int r = t & 127, h = t >> 7;
            const float4* src = (const float4*)(W + (size_t)(mBase + r) * 256 + kk + h * 16);
            float4 f0 = src[0], f1 = src[1], f2 = src[2], f3 = src[3];
            uint4 q0 = { pkbf(f0.x, f0.y), pkbf(f0.z, f0.w), pkbf(f1.x, f1.y), pkbf(f1.z, f1.w) };
            uint4 q1 = { pkbf(f2.x, f2.y), pkbf(f2.z, f2.w), pkbf(f3.x, f3.y), pkbf(f3.z, f3.w) };
            uint4* dst = (uint4*)(As + r * 32 + h * 16);
            dst[0] = q0; dst[1] = q1;
        }
        {   // stage B transposed: X[kk..+31][nBase..+127] -> Bs[n][k]
            // thread handles k-pair (k2,k2+1) x 8 tokens -> packed b32 stores
            const int k2 = (t & 15) * 2;
            const int c  = t >> 4;                  // 0..15, 8 tokens each
            const float4* s0 = (const float4*)(X + (size_t)(kk + k2) * 8192 + nBase + c * 8);
            const float4* s1 = (const float4*)(X + (size_t)(kk + k2 + 1) * 8192 + nBase + c * 8);
            float4 a0 = s0[0], a1 = s0[1], b0 = s1[0], b1 = s1[1];
            const float ar[8] = { a0.x, a0.y, a0.z, a0.w, a1.x, a1.y, a1.z, a1.w };
            const float br_[8] = { b0.x, b0.y, b0.z, b0.w, b1.x, b1.y, b1.z, b1.w };
            #pragma unroll
            for (int i = 0; i < 8; ++i)
                *(unsigned int*)(Bs + (c * 8 + i) * 32 + k2) = pkbf(ar[i], br_[i]);
        }
        __syncthreads();
        v16bf a0 = load_frag_A(As, mOff,      32, lane);
        v16bf a1 = load_frag_A(As, mOff + 16, 32, lane);
        #pragma unroll
        for (int nt = 0; nt < 4; ++nt) {
            v16bf b = load_frag_B(Bs, nOff + nt * 16, 32, lane);
            acc[0][nt] = WMMA_BF16(a0, b, acc[0][nt]);
            acc[1][nt] = WMMA_BF16(a1, b, acc[1][nt]);
        }
    }

    const int hi = lane >> 4, lo = lane & 15;
    float bias_r[2][8];
    #pragma unroll
    for (int mt = 0; mt < 2; ++mt)
        #pragma unroll
        for (int v = 0; v < 8; ++v)
            bias_r[mt][v] = bs[mBase + mOff + mt * 16 + v + hi * 8];

    // Branch hoisted OUT of the store loops -> straight-line epilogues.
    if (which == 2) {
        #pragma unroll
        for (int mt = 0; mt < 2; ++mt)
          #pragma unroll
          for (int nt = 0; nt < 4; ++nt)
            #pragma unroll
            for (int v = 0; v < 8; ++v) {
                const int m = mBase + mOff + mt * 16 + v + hi * 8;
                const int n = nBase + nOff + nt * 16 + lo;
                Vb[(size_t)(batch * 256 + m) * 8192 + n] = (__bf16)(acc[mt][nt][v] + bias_r[mt][v]);
            }
    } else {
        float* dst = (which == 1) ? Kf : Qf;
        #pragma unroll
        for (int mt = 0; mt < 2; ++mt)
          #pragma unroll
          for (int nt = 0; nt < 4; ++nt)
            #pragma unroll
            for (int v = 0; v < 8; ++v) {
                const int m = mBase + mOff + mt * 16 + v + hi * 8;
                const int n = nBase + nOff + nt * 16 + lo;
                dst[(size_t)(batch * 256 + m) * 8192 + n] = acc[mt][nt][v] + bias_r[mt][v];
            }
    }
}

// ---------------------------------------------------------------------------
// Kernel 2: softmax of K over the 8192-token axis, per row (1024 rows).
// ---------------------------------------------------------------------------
__global__ __launch_bounds__(256) void ksoftmax_kernel(
    const float* __restrict__ Kf, __bf16* __restrict__ Kb)
{
    __shared__ float red[256];
    const int row = blockIdx.x;
    const float* src = Kf + (size_t)row * 8192;
    __bf16*      dst = Kb + (size_t)row * 8192;
    const int t = threadIdx.x;

    float mx = -3.4e38f;
    for (int i = 0; i < 32; ++i) mx = fmaxf(mx, src[t + i * 256]);
    red[t] = mx; __syncthreads();
    for (int s = 128; s > 0; s >>= 1) { if (t < s) red[t] = fmaxf(red[t], red[t + s]); __syncthreads(); }
    mx = red[0]; __syncthreads();

    float sum = 0.f;
    for (int i = 0; i < 32; ++i) sum += __expf(src[t + i * 256] - mx);
    red[t] = sum; __syncthreads();
    for (int s = 128; s > 0; s >>= 1) { if (t < s) red[t] += red[t + s]; __syncthreads(); }
    const float inv = 1.0f / red[0];

    for (int i = 0; i < 32; ++i)
        dst[t + i * 256] = (__bf16)(__expf(src[t + i * 256] - mx) * inv);
}

// ---------------------------------------------------------------------------
// Kernel 3: softmax of Q over the 32 head-channels per token; writes Q-hat
// TOKEN-MAJOR [batch][head][tok][32] so GEMM B-fragments are contiguous b128s.
// ---------------------------------------------------------------------------
__global__ __launch_bounds__(256) void qsoftmax_kernel(
    const float* __restrict__ Qf, __bf16* __restrict__ Qbt)
{
    const int tid   = blockIdx.x * 256 + threadIdx.x;
    const int s     = tid & 8191;
    const int head  = (tid >> 13) & 7;
    const int batch = tid >> 16;
    const float* src = Qf + (size_t)(batch * 256 + head * 32) * 8192 + s;

    float q[32];
    float mx = -3.4e38f;
    #pragma unroll
    for (int j = 0; j < 32; ++j) { q[j] = src[(size_t)j * 8192]; mx = fmaxf(mx, q[j]); }
    float sum = 0.f;
    #pragma unroll
    for (int j = 0; j < 32; ++j) { q[j] = __expf(q[j] - mx); sum += q[j]; }
    const float inv = 1.0f / sum;

    unsigned int pk[16];
    #pragma unroll
    for (int j = 0; j < 16; ++j) pk[j] = pkbf(q[2 * j] * inv, q[2 * j + 1] * inv);
    uint4* dst = (uint4*)(Qbt + ((size_t)(batch * 8 + head) * 8192 + s) * 32);
    dst[0] = (uint4){ pk[0],  pk[1],  pk[2],  pk[3]  };
    dst[1] = (uint4){ pk[4],  pk[5],  pk[6],  pk[7]  };
    dst[2] = (uint4){ pk[8],  pk[9],  pk[10], pk[11] };
    dst[3] = (uint4){ pk[12], pk[13], pk[14], pk[15] };
}

// ---------------------------------------------------------------------------
// Kernel 4: ctx_h = K-hat_h (32x8192) @ V-hat_h^T -> 32x32, per (batch,head).
// Fragments load straight from global (rows are k-contiguous). 8 waves split
// tokens; cross-wave reduce via LDS ds_add_f32. Stores ctx^T bf16.
// ---------------------------------------------------------------------------
__global__ __launch_bounds__(256) void ctx_kernel(
    const __bf16* __restrict__ Kb, const __bf16* __restrict__ Vb,
    __bf16* __restrict__ ctxb)
{
    __shared__ float ctxs[32 * 32];
    const int t = threadIdx.x, lane = t & 31, wave = t >> 5;
    const int bh = blockIdx.x, batch = bh >> 3, h = bh & 7;
    const __bf16* Kh = Kb + (size_t)(batch * 256 + h * 32) * 8192;
    const __bf16* Vh = Vb + (size_t)(batch * 256 + h * 32) * 8192;

    v8f acc[2][2] = {};
    for (int it = 0; it < 32; ++it) {
        const int s0 = (it * 8 + wave) * 32;
        v16bf a0 = load_frag_A(Kh + s0, 0,  8192, lane);
        v16bf a1 = load_frag_A(Kh + s0, 16, 8192, lane);
        v16bf b0 = load_frag_B(Vh + s0, 0,  8192, lane);
        v16bf b1 = load_frag_B(Vh + s0, 16, 8192, lane);
        acc[0][0] = WMMA_BF16(a0, b0, acc[0][0]);
        acc[0][1] = WMMA_BF16(a0, b1, acc[0][1]);
        acc[1][0] = WMMA_BF16(a1, b0, acc[1][0]);
        acc[1][1] = WMMA_BF16(a1, b1, acc[1][1]);
    }

    for (int i = t; i < 1024; i += 256) ctxs[i] = 0.f;
    __syncthreads();
    const int hi = lane >> 4, lo = lane & 15;
    #pragma unroll
    for (int mt = 0; mt < 2; ++mt)
      #pragma unroll
      for (int nt = 0; nt < 2; ++nt)
        #pragma unroll
        for (int v = 0; v < 8; ++v)
            atomicAdd(&ctxs[(mt * 16 + v + hi * 8) * 32 + nt * 16 + lo], acc[mt][nt][v]);
    __syncthreads();

    __bf16* dst = ctxb + (size_t)bh * 1024;     // ctx^T: [v][k]
    for (int i = t; i < 1024; i += 256) {
        const int vv = i >> 5, kk = i & 31;
        dst[i] = (__bf16)ctxs[kk * 32 + vv];
    }
}

// ---------------------------------------------------------------------------
// Kernel 5: fused  agg = concat_h( ctx_h^T @ Q-hat_h[:,tile] )  (LDS bf16)
//                  rep = Wr @ agg + br + X residual, transposed store.
// grid = (tok/64, 1, batch); static LDS 48 KB.
// ---------------------------------------------------------------------------
__global__ __launch_bounds__(256) void att_rep_kernel(
    const __bf16* __restrict__ ctxb, const __bf16* __restrict__ Qbt,
    const float* __restrict__ Wr, const float* __restrict__ br,
    const float* __restrict__ x, float* __restrict__ out)
{
    __shared__ __bf16 Bs2[64 * 256];   // agg tile, token-major (k-contig)  32KB
    __shared__ __bf16 As2[256 * 32];   // Wr k-slab                         16KB

    const int t = threadIdx.x, lane = t & 31, wave = t >> 5;
    const int batch = blockIdx.z;
    const int sBase = blockIdx.x * 64;
    const int hi = lane >> 4, lo = lane & 15;

    // ---- Step A: one wave per head, att tile 32 x 64 -> Bs2 -------------
    {
        const int h = wave;
        const __bf16* ctxTh = ctxb + (size_t)(batch * 8 + h) * 1024;
        const __bf16* Qbth  = Qbt + ((size_t)(batch * 8 + h) * 8192 + sBase) * 32;
        v16bf a0 = load_frag_A(ctxTh, 0,  32, lane);
        v16bf a1 = load_frag_A(ctxTh, 16, 32, lane);
        #pragma unroll
        for (int nt = 0; nt < 4; ++nt) {
            v16bf b = load_frag_B(Qbth, nt * 16, 32, lane);
            v8f c0 = {}, c1 = {};
            c0 = WMMA_BF16(a0, b, c0);
            c1 = WMMA_BF16(a1, b, c1);
            const int tok = nt * 16 + lo;
            // 8 consecutive bf16 per accumulator half -> one ds_store_b128.
            uint4 q0 = { pkbf(c0[0], c0[1]), pkbf(c0[2], c0[3]), pkbf(c0[4], c0[5]), pkbf(c0[6], c0[7]) };
            uint4 q1 = { pkbf(c1[0], c1[1]), pkbf(c1[2], c1[3]), pkbf(c1[4], c1[5]), pkbf(c1[6], c1[7]) };
            *(uint4*)(Bs2 + tok * 256 + h * 32 + hi * 8)      = q0;
            *(uint4*)(Bs2 + tok * 256 + h * 32 + 16 + hi * 8) = q1;
        }
    }

    // ---- Step B: rep = Wr @ agg, 8 waves as 4(M)x2(N), wave 64x32 -------
    const int wm = wave & 3, wn = wave >> 2;
    v8f acc[4][2] = {};
    for (int kk = 0; kk < 256; kk += 32) {
        __syncthreads();   // first pass: fences Step-A Bs2 writes too
        {   // stage Wr[0..255][kk..+31] -> As2 (one row per thread, packed)
            const float4* src = (const float4*)(Wr + (size_t)t * 256 + kk);
            float4 f[8];
            #pragma unroll
            for (int i = 0; i < 8; ++i) f[i] = src[i];
            uint4* dst = (uint4*)(As2 + t * 32);
            #pragma unroll
            for (int i = 0; i < 4; ++i) {
                uint4 q = { pkbf(f[2*i].x, f[2*i].y), pkbf(f[2*i].z, f[2*i].w),
                            pkbf(f[2*i+1].x, f[2*i+1].y), pkbf(f[2*i+1].z, f[2*i+1].w) };
                dst[i] = q;
            }
        }
        __syncthreads();
        v16bf a[4];
        #pragma unroll
        for (int mt = 0; mt < 4; ++mt) a[mt] = load_frag_A(As2, wm * 64 + mt * 16, 32, lane);
        #pragma unroll
        for (int nt = 0; nt < 2; ++nt) {
            v16bf b = load_frag_B(Bs2 + kk, wn * 32 + nt * 16, 256, lane);
            #pragma unroll
            for (int mt = 0; mt < 4; ++mt)
                acc[mt][nt] = WMMA_BF16(a[mt], b, acc[mt][nt]);
        }
    }

    const float* Xb = x   + (size_t)batch * 256 * 8192;
    float*       ob = out + (size_t)batch * 8192 * 256;
    float brr[4][8];
    #pragma unroll
    for (int mt = 0; mt < 4; ++mt)
        #pragma unroll
        for (int v = 0; v < 8; ++v)
            brr[mt][v] = br[wm * 64 + mt * 16 + v + hi * 8];

    #pragma unroll
    for (int mt = 0; mt < 4; ++mt)
      #pragma unroll
      for (int nt = 0; nt < 2; ++nt)
        #pragma unroll
        for (int v = 0; v < 8; ++v) {
            const int m = wm * 64 + mt * 16 + v + hi * 8;          // out channel
            const int n = sBase + wn * 32 + nt * 16 + lo;          // token
            const float val = acc[mt][nt][v] + brr[mt][v] + Xb[(size_t)m * 8192 + n];
            ob[(size_t)n * 256 + m] = val;                         // transposed
        }
}

// ---------------------------------------------------------------------------
extern "C" void kernel_launch(void* const* d_in, const int* in_sizes, int n_in,
                              void* d_out, int out_size, void* d_ws, size_t ws_size,
                              hipStream_t stream)
{
    const float* x  = (const float*)d_in[0];
    const float* Wk = (const float*)d_in[1];
    const float* bk = (const float*)d_in[2];
    const float* Wq = (const float*)d_in[3];
    const float* bq = (const float*)d_in[4];
    const float* Wv = (const float*)d_in[5];
    const float* bv = (const float*)d_in[6];
    const float* Wr = (const float*)d_in[7];
    const float* br = (const float*)d_in[8];
    float* out = (float*)d_out;

    const size_t E = (size_t)4 * 256 * 8192;
    float*  Qf   = (float*)d_ws;            // 32 MB f32 Q (pre-softmax)
    float*  Kf   = Qf + E;                  // 32 MB f32 K (pre-softmax)
    __bf16* Vb   = (__bf16*)(Kf + E);       // 16 MB bf16 V
    __bf16* Kb   = Vb + E;                  // 16 MB bf16 K-hat
    __bf16* Qbt  = Kb + E;                  // 16 MB bf16 Q-hat (token-major)
    __bf16* ctxb = Qbt + E;                 // 64 KB bf16 ctx^T

    hipLaunchKernelGGL(qkv_kernel, dim3(64, 6, 4), dim3(256), 0, stream,
                       x, Wq, bq, Wk, bk, Wv, bv, Qf, Kf, Vb);
    hipLaunchKernelGGL(ksoftmax_kernel, dim3(1024), dim3(256), 0, stream, Kf, Kb);
    hipLaunchKernelGGL(qsoftmax_kernel, dim3(1024), dim3(256), 0, stream, Qf, Qbt);
    hipLaunchKernelGGL(ctx_kernel, dim3(32), dim3(256), 0, stream, Kb, Vb, ctxb);
    hipLaunchKernelGGL(att_rep_kernel, dim3(128, 1, 4), dim3(256), 0, stream,
                       ctxb, Qbt, Wr, br, x, out);
}
